// Bottleneck_66185446031552
// MI455X (gfx1250) — matrix operand order repaired and checked by reading, not compiled, not run on hardware
//
#include <hip/hip_runtime.h>

#define NPTS 50000
#define CIN  512
#define PCH  128
#define KTAPS 27
#define EPSI 1e-5f
#define NEGS 0.1f

typedef __bf16 bf16_t;
typedef __attribute__((ext_vector_type(16))) __bf16 v16bf;
typedef __attribute__((ext_vector_type(8)))  __bf16 v8bf;
typedef __attribute__((ext_vector_type(8)))  float  v8f;
typedef __attribute__((ext_vector_type(4)))  float  v4f;

union V16 { v16bf v; v8bf h[2]; bf16_t e[16]; };

// ---------------------------------------------------------------------------
// Weight prep: f32 [rows, cols] -> bf16 [cols, rows] (transpose + convert).
// blockIdx.y = batch (for W2's 27 taps).
// ---------------------------------------------------------------------------
__global__ void transpose_bf16_kernel(const float* __restrict__ src,
                                      bf16_t* __restrict__ dst,
                                      int rows, int cols) {
  const size_t batch = blockIdx.y;
  src += batch * (size_t)rows * cols;
  dst += batch * (size_t)rows * cols;
  const int i = blockIdx.x * blockDim.x + threadIdx.x;
  if (i < rows * cols) {
    const int r = i / cols, c = i % cols;
    dst[(size_t)c * rows + r] = (bf16_t)src[i];
  }
}

// ---------------------------------------------------------------------------
// GEMM1: y[N,128] = (x[N,512] @ W1[512,128]) * valid(mask)
// Wt1 is bf16 [128 out][512 in]. One 16x16 tile per wave, 8 waves cover 128 cols.
// ---------------------------------------------------------------------------
__global__ void __launch_bounds__(256) gemm1_kernel(
    const float* __restrict__ x, const float* __restrict__ mask,
    const bf16_t* __restrict__ Wt1, float* __restrict__ y) {
  const int lane = threadIdx.x & 31;
  const int wave = threadIdx.x >> 5;
  const int r  = lane & 15;
  const int kh = lane >> 4;                 // 0: lanes 0-15, 1: lanes 16-31
  const int m0 = blockIdx.x << 4;
  const float*  __restrict__ xrow = x   + (size_t)(m0 + r) * CIN;
  const bf16_t* __restrict__ wcol = Wt1 + (size_t)(wave * 16 + r) * CIN;
  v8f acc = {0.f, 0.f, 0.f, 0.f, 0.f, 0.f, 0.f, 0.f};
#pragma unroll 4
  for (int kk = 0; kk < CIN; kk += 32) {
    V16 a, b;
    const v4f a0 = *(const v4f*)(xrow + kk + kh * 8);
    const v4f a1 = *(const v4f*)(xrow + kk + kh * 8 + 4);
    const v4f a2 = *(const v4f*)(xrow + kk + 16 + kh * 8);
    const v4f a3 = *(const v4f*)(xrow + kk + 16 + kh * 8 + 4);
#pragma unroll
    for (int i = 0; i < 4; ++i) {
      a.e[i]      = (bf16_t)a0[i];
      a.e[4 + i]  = (bf16_t)a1[i];
      a.e[8 + i]  = (bf16_t)a2[i];
      a.e[12 + i] = (bf16_t)a3[i];
    }
    b.h[0] = *(const v8bf*)(wcol + kk + kh * 8);
    b.h[1] = *(const v8bf*)(wcol + kk + 16 + kh * 8);
    acc = __builtin_amdgcn_wmma_f32_16x16x32_bf16(false, a.v, false, b.v,
                                                  (short)0, acc, false, false);
  }
  const int col = wave * 16 + r;
#pragma unroll
  for (int i = 0; i < 8; ++i) {
    const int rr = m0 + i + (kh ? 8 : 0);
    const float vm = (mask[rr] > 0.f) ? 1.f : 0.f;
    y[(size_t)rr * PCH + col] = acc[i] * vm;
  }
}

// ---------------------------------------------------------------------------
// Deterministic two-stage per-channel stats (sum, sumsq) over N rows.
// ---------------------------------------------------------------------------
__global__ void stats_partial_kernel(const float* __restrict__ y,
                                     float* __restrict__ part, int C) {
  const int s = blockIdx.x;                 // stripe 0..63
  for (int c = threadIdx.x; c < C; c += blockDim.x) {
    float sm = 0.f, sq = 0.f;
    for (int r = s; r < NPTS; r += 64) {
      const float v = y[(size_t)r * C + c];
      sm += v; sq += v * v;
    }
    part[(size_t)s * 2 * C + c]     = sm;
    part[(size_t)s * 2 * C + C + c] = sq;
  }
}

__global__ void stats_final_kernel(const float* __restrict__ part,
                                   float* __restrict__ mr, int C) {
  for (int c = blockIdx.x * blockDim.x + threadIdx.x; c < C;
       c += gridDim.x * blockDim.x) {
    float sm = 0.f, sq = 0.f;
    for (int s = 0; s < 64; ++s) {
      sm += part[(size_t)s * 2 * C + c];
      sq += part[(size_t)s * 2 * C + C + c];
    }
    const float mean = sm / (float)NPTS;
    const float var  = sq / (float)NPTS - mean * mean;
    mr[c]     = mean;
    mr[C + c] = rsqrtf(var + EPSI);
  }
}

// inorm + lrelu + mask-fold + pack to bf16 (C = 128)
__global__ void elem_pack_kernel(const float* __restrict__ y,
                                 const float* __restrict__ mr,
                                 const float* __restrict__ validf, int valid_is_mask,
                                 bf16_t* __restrict__ z) {
  const int i = blockIdx.x * blockDim.x + threadIdx.x;   // exactly N*128 threads
  const int c = i & (PCH - 1);
  const int n = i >> 7;
  float v = (y[i] - mr[c]) * mr[PCH + c];
  v = (v >= 0.f) ? v : NEGS * v;
  float vm = validf[n];
  if (valid_is_mask) vm = (vm > 0.f) ? 1.f : 0.f;
  z[i] = (bf16_t)(v * vm);
}

// coverage ratio for the 27-tap partial conv
__global__ void ratio_kernel(const float* __restrict__ mask,
                             const int* __restrict__ nbr,
                             float* __restrict__ ratio,
                             float* __restrict__ valid2) {
  const int n = blockIdx.x * blockDim.x + threadIdx.x;
  if (n >= NPTS) return;
  float ms = 0.f;
#pragma unroll
  for (int k = 0; k < KTAPS; ++k)
    ms += (mask[nbr[n * KTAPS + k]] > 0.f) ? 1.f : 0.f;
  ratio[n]  = (ms > 0.f) ? ((float)KTAPS / ms) : 0.f;
  valid2[n] = (ms > 0.f) ? 1.f : 0.f;
}

// ---------------------------------------------------------------------------
// GEMM2 (gather-GEMM): y[N,128] = ratio * sum_k z1m[nbr[:,k]] @ W2t[k]
// Block: 32 points x 128 out-channels, 8 waves (one 16-col N-tile each, two
// 16-row M-subtiles sharing the B fragment). Double-buffered LDS A tile with
// register-staged gathers: tap k+1's global loads fly while tap k's WMMAs run;
// one barrier per tap.
// ---------------------------------------------------------------------------
__global__ void __launch_bounds__(256) gemm2_kernel(
    const bf16_t* __restrict__ z1m, const int* __restrict__ nbr,
    const bf16_t* __restrict__ Wt2, const float* __restrict__ ratio,
    float* __restrict__ y) {
  __shared__ __align__(16) bf16_t Abuf[2][32 * PCH];
  __shared__ int idxs[KTAPS * 32];
  const int tid = threadIdx.x;
  const int p0 = blockIdx.x << 5;           // 32 points per block
  for (int i = tid; i < KTAPS * 32; i += 256) {
    const int r = i / KTAPS, k = i % KTAPS;
    int pr = p0 + r; if (pr > NPTS - 1) pr = NPTS - 1;   // tail clamp
    idxs[k * 32 + r] = nbr[(size_t)pr * KTAPS + k];
  }
  __syncthreads();

  const int lane = tid & 31;
  const int wave = tid >> 5;
  const int r  = lane & 15;
  const int kh = lane >> 4;
  const int grow = tid >> 3;            // gather row 0..31
  const int gch  = (tid & 7) * 16;      // 16 bf16 (32B) per thread
  v8f acc0 = {0.f, 0.f, 0.f, 0.f, 0.f, 0.f, 0.f, 0.f};  // rows p0+0..15
  v8f acc1 = {0.f, 0.f, 0.f, 0.f, 0.f, 0.f, 0.f, 0.f};  // rows p0+16..31

  // prologue: stage tap 0 into registers
  v8bf ga, gb;
  {
    const bf16_t* src = z1m + (size_t)idxs[0 * 32 + grow] * PCH + gch;
    ga = *(const v8bf*)src;
    gb = *(const v8bf*)(src + 8);
  }

  for (int k = 0; k < KTAPS; ++k) {
    bf16_t* __restrict__ buf = Abuf[k & 1];
    *(v8bf*)&buf[grow * PCH + gch]     = ga;
    *(v8bf*)&buf[grow * PCH + gch + 8] = gb;
    if (k < KTAPS - 1) {                // overlap next tap's gather with compute
      const bf16_t* src = z1m + (size_t)idxs[(k + 1) * 32 + grow] * PCH + gch;
      ga = *(const v8bf*)src;
      gb = *(const v8bf*)(src + 8);
    }
    __syncthreads();                    // waits LDS stores only; loads stay in flight
    const bf16_t* __restrict__ wk = Wt2 + ((size_t)k * PCH + wave * 16 + r) * PCH;
#pragma unroll
    for (int kc = 0; kc < PCH; kc += 32) {
      V16 a0, a1, b;
      b.h[0]  = *(const v8bf*)(wk + kc + kh * 8);
      b.h[1]  = *(const v8bf*)(wk + kc + 16 + kh * 8);
      a0.h[0] = *(const v8bf*)&buf[r * PCH + kc + kh * 8];
      a0.h[1] = *(const v8bf*)&buf[r * PCH + kc + 16 + kh * 8];
      a1.h[0] = *(const v8bf*)&buf[(16 + r) * PCH + kc + kh * 8];
      a1.h[1] = *(const v8bf*)&buf[(16 + r) * PCH + kc + 16 + kh * 8];
      acc0 = __builtin_amdgcn_wmma_f32_16x16x32_bf16(false, a0.v, false, b.v,
                                                     (short)0, acc0, false, false);
      acc1 = __builtin_amdgcn_wmma_f32_16x16x32_bf16(false, a1.v, false, b.v,
                                                     (short)0, acc1, false, false);
    }
    // no second barrier: next iteration stores to the other buffer
  }

  const int col = wave * 16 + r;
#pragma unroll
  for (int i = 0; i < 8; ++i) {
    const int rr0 = p0 + i + (kh ? 8 : 0);
    const int rr1 = rr0 + 16;
    if (rr0 < NPTS) y[(size_t)rr0 * PCH + col] = acc0[i] * ratio[rr0];
    if (rr1 < NPTS) y[(size_t)rr1 * PCH + col] = acc1[i] * ratio[rr1];
  }
}

// ---------------------------------------------------------------------------
// GEMM3: out[N,512] = z2m[N,128] @ W3[128,512]  (z2m already masked)
// ---------------------------------------------------------------------------
__global__ void __launch_bounds__(256) gemm3_kernel(
    const bf16_t* __restrict__ z2m, const bf16_t* __restrict__ Wt3,
    float* __restrict__ out) {
  const int lane = threadIdx.x & 31;
  const int wave = threadIdx.x >> 5;
  const int r  = lane & 15;
  const int kh = lane >> 4;
  const int m0 = blockIdx.x << 4;
  const int n0 = blockIdx.y * 128 + wave * 16;
  const bf16_t* __restrict__ arow = z2m + (size_t)(m0 + r) * PCH;
  const bf16_t* __restrict__ wcol = Wt3 + (size_t)(n0 + r) * PCH;
  v8f acc = {0.f, 0.f, 0.f, 0.f, 0.f, 0.f, 0.f, 0.f};
#pragma unroll
  for (int kk = 0; kk < PCH; kk += 32) {
    V16 a, b;
    a.h[0] = *(const v8bf*)(arow + kk + kh * 8);
    a.h[1] = *(const v8bf*)(arow + kk + 16 + kh * 8);
    b.h[0] = *(const v8bf*)(wcol + kk + kh * 8);
    b.h[1] = *(const v8bf*)(wcol + kk + 16 + kh * 8);
    acc = __builtin_amdgcn_wmma_f32_16x16x32_bf16(false, a.v, false, b.v,
                                                  (short)0, acc, false, false);
  }
  const int col = n0 + r;
#pragma unroll
  for (int i = 0; i < 8; ++i) {
    const int rr = m0 + i + (kh ? 8 : 0);
    out[(size_t)rr * CIN + col] = acc[i];
  }
}

// final: out = lrelu(inorm(y3) + x); mout = clip(valid2 + mask, 0, 1)
__global__ void elem_final_kernel(float* __restrict__ out,
                                  const float* __restrict__ mr,
                                  const float* __restrict__ x,
                                  const float* __restrict__ mask,
                                  const float* __restrict__ valid2,
                                  float* __restrict__ mout) {
  const int i = blockIdx.x * blockDim.x + threadIdx.x;  // exactly N*512 threads
  const int c = i & (CIN - 1);
  const int n = i >> 9;
  float v = (out[i] - mr[c]) * mr[CIN + c] + x[i];
  out[i] = (v >= 0.f) ? v : NEGS * v;
  if (c == 0) {
    const float m = valid2[n] + mask[n];
    mout[n] = (m > 1.f) ? 1.f : m;
  }
}

// ---------------------------------------------------------------------------
extern "C" void kernel_launch(void* const* d_in, const int* in_sizes, int n_in,
                              void* d_out, int out_size, void* d_ws, size_t ws_size,
                              hipStream_t stream) {
  const float* x    = (const float*)d_in[0];
  const float* mask = (const float*)d_in[1];
  const int*   nbr  = (const int*)  d_in[2];
  const float* W1   = (const float*)d_in[3];
  const float* W2   = (const float*)d_in[4];
  const float* W3   = (const float*)d_in[5];
  float* out  = (float*)d_out;                   // [N,512]
  float* mout = out + (size_t)NPTS * CIN;        // [N]

  char* ws = (char*)d_ws;
  size_t off = 0;
  auto alloc = [&](size_t bytes) -> void* {
    off = (off + 255) & ~(size_t)255;
    void* p = ws + off;
    off += bytes;
    return p;
  };
  bf16_t* Wt1   = (bf16_t*)alloc((size_t)CIN * PCH * 2);
  bf16_t* Wt2   = (bf16_t*)alloc((size_t)KTAPS * PCH * PCH * 2);
  bf16_t* Wt3   = (bf16_t*)alloc((size_t)PCH * CIN * 2);
  float*  yb    = (float*) alloc((size_t)NPTS * PCH * 4);
  bf16_t* z1m   = (bf16_t*)alloc((size_t)NPTS * PCH * 2);
  bf16_t* z2m   = (bf16_t*)alloc((size_t)NPTS * PCH * 2);
  float*  ratio = (float*) alloc((size_t)NPTS * 4);
  float*  valid2= (float*) alloc((size_t)NPTS * 4);
  float*  part  = (float*) alloc((size_t)64 * 2 * CIN * 4);
  float*  mr    = (float*) alloc((size_t)2 * CIN * 4);
  (void)ws_size; (void)in_sizes; (void)n_in; (void)out_size;

  // weight prep (transpose + bf16)
  transpose_bf16_kernel<<<dim3(256, 1), 256, 0, stream>>>(W1, Wt1, CIN, PCH);
  transpose_bf16_kernel<<<dim3(64, KTAPS), 256, 0, stream>>>(W2, Wt2, PCH, PCH);
  transpose_bf16_kernel<<<dim3(256, 1), 256, 0, stream>>>(W3, Wt3, PCH, CIN);

  // stage 1: pconv_k1 + inorm + lrelu
  gemm1_kernel<<<NPTS / 16, 256, 0, stream>>>(x, mask, Wt1, yb);
  stats_partial_kernel<<<64, 256, 0, stream>>>(yb, part, PCH);
  stats_final_kernel<<<1, 256, 0, stream>>>(part, mr, PCH);
  elem_pack_kernel<<<(NPTS * PCH) / 256, 256, 0, stream>>>(yb, mr, mask, 1, z1m);

  // stage 2: 27-tap gather partial conv + inorm + lrelu
  ratio_kernel<<<(NPTS + 255) / 256, 256, 0, stream>>>(mask, nbr, ratio, valid2);
  gemm2_kernel<<<(NPTS + 31) / 32, 256, 0, stream>>>(z1m, nbr, Wt2, ratio, yb);
  stats_partial_kernel<<<64, 256, 0, stream>>>(yb, part, PCH);
  stats_final_kernel<<<1, 256, 0, stream>>>(part, mr, PCH);
  elem_pack_kernel<<<(NPTS * PCH) / 256, 256, 0, stream>>>(yb, mr, valid2, 0, z2m);

  // stage 3: pconv_k1 expand + inorm + residual lrelu + mask out
  gemm3_kernel<<<dim3(NPTS / 16, 4), 256, 0, stream>>>(z2m, Wt3, out);
  stats_partial_kernel<<<64, 256, 0, stream>>>(out, part, CIN);
  stats_final_kernel<<<2, 256, 0, stream>>>(part, mr, CIN);
  elem_final_kernel<<<(NPTS * CIN) / 256, 256, 0, stream>>>(out, mr, x, mask,
                                                            valid2, mout);
}
